// NeuralWavefunction_80229989089987
// MI455X (gfx1250) — matrix operand order repaired and checked by reading, not compiled, not run on hardware
//
#include <hip/hip_runtime.h>
#include <hip/hip_bf16.h>
#include <hip/hip_fp16.h>

// ---------------------------------------------------------------------------
// NeuralWavefunction forward for MI455X (gfx1250, wave32, WMMA).
//
// All GEMMs run on v_wmma_f32_16x16x32_f16 (f16 in, f32 accumulate).  The
// Mamba block is fully fused: one 128-thread workgroup per (walker, electron)
// sequence keeps every intermediate (X, si, xi/z, xc, xd, scan state) in
// LDS/registers, so the 240 MB of Mamba activations never touch HBM; the
// kernel is bounded by pair_h traffic (~64 MB * few passes) at 23.3 TB/s.
// GEMM weights are pre-converted to f16 once per launch (halves the per-block
// weight staging traffic and removes per-block convert VALU); staging copies
// are uint4 (b128) moves.  global_prefetch warms the next phase's weights.
// Recurrent scan + exp-heavy parts stay on f32 VALU.
//
// Input pointer order assumed (setup_inputs() dict insertion order, params
// pytree in insertion order, each linear as {w, b}):
//   0 r_electrons [1024*16*3]   1 r_nuclei [12]   2 charges [4]
//   3 spin_mask_parallel [256] (int)
//   4..11: se1.w, se1.b, se2.w, se2.b, pe1.w, pe1.b, pe2.w, pe2.b
//   then per layer (x3), 24 arrays:
//   s1.w,s1.b, s2.w,s2.b, p1.w,p1.b, p2.w,p2.b, proj.w,proj.b,
//   m_in.w,m_in.b, m_x.w,m_x.b, m_dt.w,m_dt.b, A_log, Dp, m_out.w,m_out.b,
//   ln_s_g, ln_s_b, ln_p_g, ln_p_b
// ---------------------------------------------------------------------------

#define NW   1024
#define NE   16
#define NN   4
#define DM   64
#define DI   128
#define DSTATE 16
#define LSEQ 15

// f16 weight arena per layer (element offsets)
#define OFF_PROJ 0          // 64x128   = 8192
#define OFF_MIN  8192       // 256x64   = 16384
#define OFF_MX   24576      // 48x128   = 6144 (36 valid rows, zero-padded)
#define OFF_P1   30720      // 64x64
#define OFF_P2   34816
#define OFF_S1   38912
#define OFF_S2   43008
#define WH_PER_LAYER 47104

typedef __attribute__((ext_vector_type(16))) _Float16 v16h;
typedef __attribute__((ext_vector_type(8)))  float    v8f;

static __device__ __forceinline__ float silu_f(float x) {
  return x / (1.f + expf(-x));
}

static __device__ __forceinline__ v8f wmma16(v16h a, v16h b, v8f c) {
  // D = A(16x32 f16) * B(32x16 f16) + C(16x16 f32)
  return __builtin_amdgcn_wmma_f32_16x16x32_f16(
      /*neg_a=*/false, a, /*neg_b=*/false, b,
      /*c_mod=*/(short)0, c, /*reuse_a=*/false, /*reuse_b=*/false);
}

// A fragment from an LDS f16 tile [16][ld] (row-major), K-window [k0, k0+32).
static __device__ __forceinline__ v16h frag_a(const _Float16* t, int ld, int k0) {
  int lane = threadIdx.x & 31;
  int m = lane & 15, hh = lane >> 4;
  const _Float16* p0 = t + m * ld + k0 + 16 * hh;      // K = k0+16h .. +15
  v16h a;
#pragma unroll
  for (int e = 0; e < 16; ++e) a[e] = p0[e];
  return a;
}

// B fragment from LDS f16 weights W[N][K] row-major: B[k][n] = W[n][k].
static __device__ __forceinline__ v16h frag_b(const _Float16* w, int K, int n0, int k0) {
  int lane = threadIdx.x & 31;
  int n = n0 + (lane & 15), hh = lane >> 4;
  const _Float16* p = w + n * K + k0 + 16 * hh;        // K = k0+16h .. +15
  v16h b;
#pragma unroll
  for (int e = 0; e < 16; ++e) b[e] = p[e];
  return b;
}

// b128 LDS staging copy (both sides 16B aligned).
static __device__ __forceinline__ void copy16(_Float16* dst, const _Float16* src,
                                              int halves, int tid, int nthr) {
  uint4* d4 = (uint4*)dst;
  const uint4* s4 = (const uint4*)src;
  for (int i = tid; i < (halves >> 3); i += nthr) d4[i] = s4[i];
}

// ---------------------------------------------------------------------------
// One-time f32 -> f16 conversion of all WMMA weights for one layer.
// m_x is zero-padded from 36 to 48 output rows.
// ---------------------------------------------------------------------------
__global__ __launch_bounds__(256) void convert_w_kernel(
    const float* __restrict__ projw, const float* __restrict__ minw,
    const float* __restrict__ mxw,   const float* __restrict__ p1w,
    const float* __restrict__ p2w,   const float* __restrict__ s1w,
    const float* __restrict__ s2w,   _Float16* __restrict__ dst) {
  int i = blockIdx.x * blockDim.x + threadIdx.x;
  if (i >= WH_PER_LAYER) return;
  float v;
  if      (i < OFF_MIN)  v = projw[i - OFF_PROJ];
  else if (i < OFF_MX)   v = minw[i - OFF_MIN];
  else if (i < OFF_P1) { int j = i - OFF_MX; v = (j < 36 * 128) ? mxw[j] : 0.f; }
  else if (i < OFF_P2)   v = p1w[i - OFF_P1];
  else if (i < OFF_S1)   v = p2w[i - OFF_P2];
  else if (i < OFF_S2)   v = s1w[i - OFF_S1];
  else                   v = s2w[i - OFF_S2];
  dst[i] = (_Float16)v;
}

// ---------------------------------------------------------------------------
// Electron embedding: h = se2( tanh( se1( [r_e, r_en, r_en^2, exp(-q r_en)] )))
// one 64-thread block per (w,e) row.
// ---------------------------------------------------------------------------
__global__ __launch_bounds__(64) void embed_h_kernel(
    const float* __restrict__ re, const float* __restrict__ rn,
    const float* __restrict__ q,
    const float* __restrict__ w1, const float* __restrict__ b1,
    const float* __restrict__ w2, const float* __restrict__ b2,
    float* __restrict__ h) {
  int row = blockIdx.x;           // w*NE + e
  int o   = threadIdx.x;          // 0..63
  __shared__ float t1[DM];
  float f[15];
  float ex = re[row * 3 + 0], ey = re[row * 3 + 1], ez = re[row * 3 + 2];
  f[0] = ex; f[1] = ey; f[2] = ez;
#pragma unroll
  for (int n = 0; n < NN; ++n) {
    float dx = ex - rn[n * 3 + 0], dy = ey - rn[n * 3 + 1], dz = ez - rn[n * 3 + 2];
    float d2 = dx * dx + dy * dy + dz * dz;
    float r = d2 > 0.f ? sqrtf(d2) : 0.f;
    f[3 + n] = r; f[7 + n] = r * r; f[11 + n] = expf(-q[n] * r);
  }
  float s = b1[o];
#pragma unroll
  for (int i = 0; i < 15; ++i) s += f[i] * w1[o * 15 + i];
  t1[o] = tanhf(s);
  __syncthreads();
  float s2 = b2[o];
  for (int i = 0; i < DM; ++i) s2 += t1[i] * w2[o * DM + i];
  h[row * DM + o] = s2;
}

// ---------------------------------------------------------------------------
// Pair embedding + r_ee.  One 64-thread block per (w,e,f) row.
// ---------------------------------------------------------------------------
__global__ __launch_bounds__(64) void embed_pair_kernel(
    const float* __restrict__ re, const int* __restrict__ spin,
    const float* __restrict__ w1, const float* __restrict__ b1,
    const float* __restrict__ w2, const float* __restrict__ b2,
    float* __restrict__ pair_h, float* __restrict__ r_ee) {
  int p = blockIdx.x;             // w*256 + e*16 + f
  int o = threadIdx.x;
  int w = p >> 8, e = (p >> 4) & 15, fe = p & 15;
  __shared__ float t1[DM];
  float ax = re[(w * NE + e) * 3 + 0], ay = re[(w * NE + e) * 3 + 1], az = re[(w * NE + e) * 3 + 2];
  float vx = ax - re[(w * NE + fe) * 3 + 0];
  float vy = ay - re[(w * NE + fe) * 3 + 1];
  float vz = az - re[(w * NE + fe) * 3 + 2];
  float d2 = vx * vx + vy * vy + vz * vz;
  float r = d2 > 0.f ? sqrtf(d2) : 0.f;
  if (o == 0) r_ee[p] = r;
  float pf[7] = { r, r * r, expf(-r), vx, vy, vz, spin[e * NE + fe] ? 1.f : 0.f };
  float s = b1[o];
#pragma unroll
  for (int i = 0; i < 7; ++i) s += pf[i] * w1[o * 7 + i];
  t1[o] = tanhf(s);
  __syncthreads();
  float s2 = b2[o];
  for (int i = 0; i < DM; ++i) s2 += t1[i] * w2[o * DM + i];
  pair_h[p * DM + o] = s2;
}

// ---------------------------------------------------------------------------
// Stable ascending argsort of 16 distances (diagonal = inf), keep 15.
// ---------------------------------------------------------------------------
__global__ __launch_bounds__(256) void argsort_kernel(
    const float* __restrict__ r_ee, int* __restrict__ idx15) {
  int gid = blockIdx.x * blockDim.x + threadIdx.x;
  if (gid >= NW * NE) return;
  int w = gid / NE, e = gid % NE;
  float d[NE];
#pragma unroll
  for (int j = 0; j < NE; ++j) d[j] = r_ee[(w * NE + e) * NE + j];
  d[e] = __builtin_inff();
#pragma unroll
  for (int k = 0; k < LSEQ; ++k) {
    int best = 0; float bv = d[0];
#pragma unroll
    for (int j = 1; j < NE; ++j) if (d[j] < bv) { bv = d[j]; best = j; }  // first-min => stable
    idx15[gid * LSEQ + k] = best;
    d[best] = __builtin_inff();
  }
}

// ---------------------------------------------------------------------------
// Fused ssm_agg: one 128-thread block (4 waves) per sequence (w,e).
// gather -> proj(WMMA) -> m_in(WMMA) -> conv+silu -> m_x(WMMA) -> dt/scan ->
// gate -> mean -> m_out matvec.  ~63 KB LDS.
// wh = pre-converted f16 weight arena for this layer.
// ---------------------------------------------------------------------------
__global__ __launch_bounds__(128) void agg_kernel(
    const float* __restrict__ h, const float* __restrict__ pair_h,
    const int* __restrict__ idx15, const _Float16* __restrict__ wh,
    const float* __restrict__ projb, const float* __restrict__ minb,
    const float* __restrict__ mxb,
    const float* __restrict__ mdtw,  const float* __restrict__ mdtb,
    const float* __restrict__ alog,  const float* __restrict__ dp,
    const float* __restrict__ moutw, const float* __restrict__ moutb,
    float* __restrict__ pair_agg) {
  __shared__ __align__(16) _Float16 wbuf[256 * 64];   // 32 KB, reused per GEMM
  __shared__ __align__(16) _Float16 Xh[16 * 128];     // proj input [t][concat(h_nb,p_nb)]
  __shared__ __align__(16) _Float16 si[16 * 64];      // silu(proj) -> m_in input
  __shared__ float    xif[16 * 128];    // xi (pre-conv)
  __shared__ float    zf [16 * 128];    // z  (gate)
  __shared__ __align__(16) _Float16 xc16[16 * 128];   // silu(conv(xi)) -> m_x input
  __shared__ float    xd[16 * 48];      // [dt_r(4) | B(16) | C(16) | pad(12)]
  __shared__ float    accv[128];        // mean_t of gated output
  __shared__ int      nidx[16];

  int tid = threadIdx.x, lane = tid & 31, wv = tid >> 5;
  int seq = blockIdx.x;                 // w*NE + e
  int w = seq >> 4;

  // warm later-phase weights while proj runs (global_prefetch_b8)
  __builtin_prefetch(wh + OFF_MIN + tid * 128, 0, 1);   // 32 KB m_in
  __builtin_prefetch(wh + OFF_MX + tid * 48, 0, 1);     // 12 KB m_x

  if (tid < LSEQ) nidx[tid] = idx15[seq * LSEQ + tid];
  copy16(wbuf, wh + OFF_PROJ, DM * 2 * DM, tid, 128);   // 64x128 proj weights
  __syncthreads();

  // Gather X[t] = concat(h[w, nb_t], pair_h[w, e, nb_t]); row 15 = 0.
  for (int i = tid; i < LSEQ * 128; i += 128) {
    int t = i >> 7, c = i & 127;
    int n = nidx[t];
    float v = (c < 64) ? h[(w * NE + n) * DM + c]
                       : pair_h[(seq * NE + n) * DM + (c - 64)];
    Xh[t * 128 + c] = (_Float16)v;
  }
  Xh[LSEQ * 128 + tid] = (_Float16)0.f;
  __syncthreads();

  // ---- proj: M16 K128 N64, wave wv handles n-tile wv -------------------
  {
    v8f acc = {};
#pragma unroll
    for (int k0 = 0; k0 < 128; k0 += 32) {
      v16h a = frag_a(Xh, 128, k0);
      acc = wmma16(a, frag_b(wbuf, 128, wv * 16, k0), acc);
    }
    int n = lane & 15, hh = lane >> 4;
#pragma unroll
    for (int r = 0; r < 8; ++r) {
      int rr = r + 8 * hh;
      float v = acc[r] + projb[wv * 16 + n];
      si[rr * 64 + wv * 16 + n] = (_Float16)silu_f(v);
    }
  }
  __syncthreads();

  copy16(wbuf, wh + OFF_MIN, 256 * 64, tid, 128);       // 256x64 m_in weights
  __syncthreads();

  // ---- m_in: M16 K64 N256 -> xi | z ------------------------------------
  for (int nti = 0; nti < 4; ++nti) {
    int nt = wv * 4 + nti;
    v8f acc = {};
#pragma unroll
    for (int k0 = 0; k0 < 64; k0 += 32) {
      v16h a = frag_a(si, 64, k0);
      acc = wmma16(a, frag_b(wbuf, 64, nt * 16, k0), acc);
    }
    int n = lane & 15, hh = lane >> 4;
    int col = nt * 16 + n;
#pragma unroll
    for (int r = 0; r < 8; ++r) {
      int rr = r + 8 * hh;
      float v = acc[r] + minb[col];
      if (col < 128) xif[rr * 128 + col] = v;
      else           zf[rr * 128 + col - 128] = v;
    }
  }
  __syncthreads();

  // ---- causal conv(k=3, 1/3) + silu; per-channel state prep ------------
  int ch = tid;                         // 0..127 == DI
  float eAr[DSTATE];
  float dpv = dp[ch];
  {
    float p1 = 0.f, p2 = 0.f;
    for (int t = 0; t < LSEQ; ++t) {
      float cur = xif[t * 128 + ch];
      float v = (cur + p1 + p2) * (1.f / 3.f);
      xc16[t * 128 + ch] = (_Float16)silu_f(v);
      p2 = p1; p1 = cur;
    }
    xc16[LSEQ * 128 + ch] = (_Float16)0.f;
#pragma unroll
    for (int s = 0; s < DSTATE; ++s) eAr[s] = expf(alog[ch * DSTATE + s]);
  }
  __syncthreads();

  copy16(wbuf, wh + OFF_MX, 48 * 128, tid, 128);        // padded 48x128 m_x weights
  __syncthreads();

  // ---- m_x: M16 K128 N48 (waves 0..2) ----------------------------------
  if (wv < 3) {
    v8f acc = {};
#pragma unroll
    for (int k0 = 0; k0 < 128; k0 += 32) {
      v16h a = frag_a(xc16, 128, k0);
      acc = wmma16(a, frag_b(wbuf, 128, wv * 16, k0), acc);
    }
    int n = lane & 15, hh = lane >> 4;
    int col = wv * 16 + n;
    float bb = (col < 36) ? mxb[col] : 0.f;
#pragma unroll
    for (int r = 0; r < 8; ++r) {
      int rr = r + 8 * hh;
      xd[rr * 48 + col] = acc[r] + bb;
    }
  }
  __syncthreads();

  // ---- dt = softplus(m_dt(dt_r)); selective scan; gate; mean -----------
  {
    float hst[DSTATE];
#pragma unroll
    for (int s = 0; s < DSTATE; ++s) hst[s] = 0.f;
    float mb = mdtb[ch];
    float w0 = mdtw[ch * 4 + 0], w1 = mdtw[ch * 4 + 1];
    float w2 = mdtw[ch * 4 + 2], w3 = mdtw[ch * 4 + 3];
    float acc = 0.f;
    for (int t = 0; t < LSEQ; ++t) {
      const float* xdt = xd + t * 48;
      float dx = mb + xdt[0] * w0 + xdt[1] * w1 + xdt[2] * w2 + xdt[3] * w3;
      float dtv = fmaxf(dx, 0.f) + log1pf(expf(-fabsf(dx)));   // softplus
      float xcv = (float)xc16[t * 128 + ch];
      float y = 0.f;
#pragma unroll
      for (int s = 0; s < DSTATE; ++s) {
        float dA = expf(-eAr[s] * dtv);                        // exp(A*dt), A=-exp(A_log)
        hst[s] = dA * hst[s] + dtv * xdt[4 + s] * xcv;         // + dt*B*x
        y += hst[s] * xdt[20 + s];                             // * C
      }
      float zv = zf[t * 128 + ch];
      acc += (y + xcv * dpv) * silu_f(zv);
    }
    accv[ch] = acc * (1.f / LSEQ);                             // mean over t
  }
  __syncthreads();

  // ---- m_out matvec (mean commutes with the linear) --------------------
  if (tid < DM) {
    float s = moutb[tid];
    for (int i = 0; i < DI; ++i) s += accv[i] * moutw[tid * DI + i];
    pair_agg[seq * DM + tid] = s;
  }
}

// ---------------------------------------------------------------------------
// out = LayerNorm(x + W2*silu(W1*x + b1) + b2 [+ extra]) over rows of 64.
// 128 threads = 4 waves, 16 rows/wave, WMMA for both 64x64 GEMMs.
// NOTE: called with out == x (in place): x/out intentionally NOT restrict.
// ---------------------------------------------------------------------------
__global__ __launch_bounds__(128) void mlp_ln_kernel(
    const float* x, const float* extra, float* out,
    const _Float16* __restrict__ w1, const float* __restrict__ b1,
    const _Float16* __restrict__ w2, const float* __restrict__ b2,
    const float* __restrict__ gamma, const float* __restrict__ beta) {
  __shared__ __align__(16) _Float16 w1h[DM * DM];
  __shared__ __align__(16) _Float16 w2h[DM * DM];
  __shared__ __align__(16) _Float16 xt[4][16 * DM];
  __shared__ __align__(16) _Float16 mt[4][16 * DM];
  __shared__ float    yt[4][16 * DM];
  int tid = threadIdx.x, lane = tid & 31, wv = tid >> 5;
  copy16(w1h, w1, DM * DM, tid, 128);
  copy16(w2h, w2, DM * DM, tid, 128);
  int row0 = blockIdx.x * 64 + wv * 16;
  for (int i = lane; i < 16 * DM; i += 32) xt[wv][i] = (_Float16)x[row0 * DM + i];
  __syncthreads();

  int n = lane & 15, hh = lane >> 4;
  // GEMM1 + silu -> mt
  {
    v8f acc[4] = {};
#pragma unroll
    for (int k0 = 0; k0 < DM; k0 += 32) {
      v16h a = frag_a(xt[wv], DM, k0);
#pragma unroll
      for (int nt = 0; nt < 4; ++nt) acc[nt] = wmma16(a, frag_b(w1h, DM, nt * 16, k0), acc[nt]);
    }
#pragma unroll
    for (int nt = 0; nt < 4; ++nt)
#pragma unroll
      for (int r = 0; r < 8; ++r) {
        int rr = r + 8 * hh;
        float v = acc[nt][r] + b1[nt * 16 + n];
        mt[wv][rr * DM + nt * 16 + n] = (_Float16)silu_f(v);
      }
  }
  __syncthreads();
  // GEMM2 -> yt
  {
    v8f acc[4] = {};
#pragma unroll
    for (int k0 = 0; k0 < DM; k0 += 32) {
      v16h a = frag_a(mt[wv], DM, k0);
#pragma unroll
      for (int nt = 0; nt < 4; ++nt) acc[nt] = wmma16(a, frag_b(w2h, DM, nt * 16, k0), acc[nt]);
    }
#pragma unroll
    for (int nt = 0; nt < 4; ++nt)
#pragma unroll
      for (int r = 0; r < 8; ++r) {
        int rr = r + 8 * hh;
        yt[wv][rr * DM + nt * 16 + n] = acc[nt][r] + b2[nt * 16 + n];
      }
  }
  __syncthreads();
  // Residual (+extra) and LayerNorm; one lane per row.
  if (lane < 16) {
    int grow = row0 + lane;
    float s = 0.f;
    for (int i = 0; i < DM; ++i) {
      float t = x[grow * DM + i] + yt[wv][lane * DM + i];
      if (extra) t += extra[grow * DM + i];
      yt[wv][lane * DM + i] = t;
      s += t;
    }
    float mean = s * (1.f / DM);
    float vs = 0.f;
    for (int i = 0; i < DM; ++i) { float d = yt[wv][lane * DM + i] - mean; vs += d * d; }
    float inv = rsqrtf(vs * (1.f / DM) + 1e-5f);
    for (int i = 0; i < DM; ++i)
      out[grow * DM + i] = (yt[wv][lane * DM + i] - mean) * inv * gamma[i] + beta[i];
  }
}

// ---------------------------------------------------------------------------
extern "C" void kernel_launch(void* const* d_in, const int* in_sizes, int n_in,
                              void* d_out, int out_size, void* d_ws, size_t ws_size,
                              hipStream_t stream) {
  (void)in_sizes; (void)n_in; (void)out_size; (void)ws_size;
  const float* r_e  = (const float*)d_in[0];
  const float* r_n  = (const float*)d_in[1];
  const float* q    = (const float*)d_in[2];
  const int*   spin = (const int*)d_in[3];
  int p = 4;
  auto nxt = [&]() { return (const float*)d_in[p++]; };
  const float *se1w = nxt(), *se1b = nxt(), *se2w = nxt(), *se2b = nxt();
  const float *pe1w = nxt(), *pe1b = nxt(), *pe2w = nxt(), *pe2b = nxt();
  struct LP {
    const float *s1w,*s1b,*s2w,*s2b,*p1w,*p1b,*p2w,*p2b,*projw,*projb,
                *minw,*minb,*mxw,*mxb,*mdtw,*mdtb,*alog,*dp,*moutw,*moutb,
                *lnsg,*lnsb,*lnpg,*lnpb;
  } L[3];
  for (int l = 0; l < 3; ++l) {
    L[l].s1w = nxt(); L[l].s1b = nxt(); L[l].s2w = nxt(); L[l].s2b = nxt();
    L[l].p1w = nxt(); L[l].p1b = nxt(); L[l].p2w = nxt(); L[l].p2b = nxt();
    L[l].projw = nxt(); L[l].projb = nxt();
    L[l].minw  = nxt(); L[l].minb  = nxt();
    L[l].mxw   = nxt(); L[l].mxb   = nxt();
    L[l].mdtw  = nxt(); L[l].mdtb  = nxt();
    L[l].alog  = nxt(); L[l].dp    = nxt();
    L[l].moutw = nxt(); L[l].moutb = nxt();
    L[l].lnsg = nxt(); L[l].lnsb = nxt(); L[l].lnpg = nxt(); L[l].lnpb = nxt();
  }

  // Workspace layout (~77.2 MB)
  float* hbuf  = (float*)d_ws;                       // [NW*NE*DM]
  float* pairh = hbuf  + NW * NE * DM;               // [NW*NE*NE*DM]
  float* ree   = pairh + NW * NE * NE * DM;          // [NW*NE*NE]
  float* pagg  = ree   + NW * NE * NE;               // [NW*NE*DM]
  int*   idx   = (int*)(pagg + NW * NE * DM);        // [NW*NE*LSEQ]
  _Float16* wh = (_Float16*)(idx + NW * NE * LSEQ);  // [3 * WH_PER_LAYER] f16

  for (int l = 0; l < 3; ++l)
    convert_w_kernel<<<(WH_PER_LAYER + 255) / 256, 256, 0, stream>>>(
        L[l].projw, L[l].minw, L[l].mxw, L[l].p1w, L[l].p2w, L[l].s1w, L[l].s2w,
        wh + (size_t)l * WH_PER_LAYER);

  embed_h_kernel<<<NW * NE, 64, 0, stream>>>(r_e, r_n, q, se1w, se1b, se2w, se2b, hbuf);
  embed_pair_kernel<<<NW * NE * NE, 64, 0, stream>>>(r_e, spin, pe1w, pe1b, pe2w, pe2b, pairh, ree);
  argsort_kernel<<<(NW * NE + 255) / 256, 256, 0, stream>>>(ree, idx);

  for (int l = 0; l < 3; ++l) {
    const _Float16* whl = wh + (size_t)l * WH_PER_LAYER;
    agg_kernel<<<NW * NE, 128, 0, stream>>>(hbuf, pairh, idx, whl,
        L[l].projb, L[l].minb, L[l].mxb,
        L[l].mdtw, L[l].mdtb, L[l].alog, L[l].dp, L[l].moutw, L[l].moutb, pagg);
    mlp_ln_kernel<<<NW * NE * NE / 64, 128, 0, stream>>>(pairh, nullptr, pairh,
        whl + OFF_P1, L[l].p1b, whl + OFF_P2, L[l].p2b, L[l].lnpg, L[l].lnpb);
    mlp_ln_kernel<<<NW * NE / 64, 128, 0, stream>>>(hbuf, pagg, hbuf,
        whl + OFF_S1, L[l].s1b, whl + OFF_S2, L[l].s2b, L[l].lnsg, L[l].lnsb);
  }

  hipMemcpyAsync(d_out, hbuf, (size_t)NW * NE * DM * sizeof(float),
                 hipMemcpyDeviceToDevice, stream);
}
